// SpeedySpeech_4475355922920
// MI455X (gfx1250) — matrix operand order repaired and checked by reading, not compiled, not run on hardware
//
#include <hip/hip_runtime.h>

typedef __attribute__((ext_vector_type(16))) _Float16 v16h;
typedef __attribute__((ext_vector_type(8)))  _Float16 v8h;
typedef __attribute__((ext_vector_type(8)))  float    v8f;

#define CCH 128

// ---------------- small scalar kernels ----------------

__global__ void gather_emb_k(const int* __restrict__ ph, const float* __restrict__ emb,
                             _Float16* __restrict__ out, int rows) {
  int i = blockIdx.x * blockDim.x + threadIdx.x;
  if (i >= rows * CCH) return;
  int row = i >> 7, c = i & 127;
  out[i] = (_Float16)emb[ph[row] * CCH + c];
}

// per-channel mean / rsqrt(var+eps) over all rows (train-mode BN over (B,T))
__global__ void bn_stats_k(const float* __restrict__ x, float* __restrict__ stats,
                           int rows, int relu_first) {
  int c = blockIdx.x, tid = threadIdx.x;
  float s = 0.f, sq = 0.f;
  for (int r = tid; r < rows; r += blockDim.x) {
    float v = x[(size_t)r * CCH + c];
    if (relu_first) v = fmaxf(v, 0.f);
    s += v; sq += v * v;
  }
  __shared__ float ss[256], s2[256];
  ss[tid] = s; s2[tid] = sq; __syncthreads();
  for (int off = 128; off; off >>= 1) {
    if (tid < off) { ss[tid] += ss[tid + off]; s2[tid] += s2[tid + off]; }
    __syncthreads();
  }
  if (tid == 0) {
    float m = ss[0] / rows;
    float var = s2[0] / rows - m * m;
    stats[c] = m;
    stats[CCH + c] = rsqrtf(var + 1e-5f);
  }
}

// mode 0: f16(relu(bn(x)))   mode 1: f16(bn(relu(x)))   mode 2: f16(x)
__global__ void bnrelu_k(const float* __restrict__ x, const float* __restrict__ stats,
                         const float* __restrict__ g, const float* __restrict__ be,
                         _Float16* __restrict__ y, int n, int mode) {
  int i = blockIdx.x * blockDim.x + threadIdx.x;
  if (i >= n) return;
  float v = x[i];
  if (mode == 2) { y[i] = (_Float16)v; return; }
  int c = i & 127;
  if (mode == 1) v = fmaxf(v, 0.f);
  v = (v - stats[c]) * stats[CCH + c] * g[c] + be[c];
  if (mode == 0) v = fmaxf(v, 0.f);
  y[i] = (_Float16)v;
}

// repack f32 weights [KW, Cin, Cout] into per-wave B-fragment layout:
// frag (kw, kc, nt): 32 lanes x 16 halves; lane -> (lo=col, hi=K half), half h -> K = hi*16+h
__global__ void pack_w_k(const float* __restrict__ w, _Float16* __restrict__ wp,
                         int KW, int Cin, int Cout) {
  int idx = blockIdx.x * blockDim.x + threadIdx.x;
  int total = KW * (Cin >> 5) * (Cout >> 4) * 512;
  if (idx >= total) return;
  int h = idx & 15;
  int lane = (idx >> 4) & 31;
  int frag = idx >> 9;
  int nt = frag % (Cout >> 4);
  int rest = frag / (Cout >> 4);
  int kc = rest % (Cin >> 5);
  int kw = rest / (Cin >> 5);
  int lo = lane & 15, hi = lane >> 4;
  int kin = kc * 32 + hi * 16 + h;
  int co = nt * 16 + lo;
  wp[idx] = (_Float16)w[((size_t)(kw * Cin + kin)) * Cout + co];
}

// ---------------- WMMA implicit-GEMM dilated conv1d ----------------
// a: [B*T, 128] f16 (already BN/ReLU'd); one wave computes MT 16x16 tiles sharing
// each B (weight) fragment; T = 1<<LT so batch/time split is shift/mask.
template <int MT, int LT>
__global__ void conv_wmma_k(const _Float16* __restrict__ a, const _Float16* __restrict__ wp,
                            const float* __restrict__ bias, const float* __restrict__ resid,
                            float* __restrict__ out, int Cout,
                            int KW, int dil, int pl, int relu) {
  constexpr int T = 1 << LT;
  constexpr int KCN = CCH / 32;   // 4 K-chunks of 32
  int lane = threadIdx.x & 31;
  int wv = threadIdx.x >> 5;
  int lo = lane & 15, hi = lane >> 4;
  int ntiles = Cout >> 4;
  int tileN = blockIdx.y * 8 + wv;
  if (tileN >= ntiles) return;
  int tileM0 = blockIdx.x * MT;

  v8f acc[MT];
#pragma unroll
  for (int mt = 0; mt < MT; ++mt) { v8f z = {}; acc[mt] = z; }

  for (int kw = 0; kw < KW; ++kw) {
    const _Float16* arow[MT];
    bool inb[MT];
#pragma unroll
    for (int mt = 0; mt < MT; ++mt) {
      int m = (tileM0 + mt) * 16 + lo;             // A row (same for both K halves)
      int bb = m >> LT;
      int t = m & (T - 1);
      int tin = t + kw * dil - pl;
      inb[mt] = (tin >= 0) && (tin < T);
      arow[mt] = a + (((size_t)bb << LT) + (size_t)(inb[mt] ? tin : 0)) * CCH;
    }
#pragma unroll
    for (int kc = 0; kc < KCN; ++kc) {
      const _Float16* bp = wp + ((size_t)(kw * KCN + kc) * ntiles + tileN) * 512 + lane * 16;
      v16h bv = *(const v16h*)bp;                  // one 32B load feeds MT WMMAs
#pragma unroll
      for (int mt = 0; mt < MT; ++mt) {
        v16h av;
        if (inb[mt]) {
          v8h l8 = *(const v8h*)(arow[mt] + kc * 32 + hi * 16);      // K = hi*16 + 0..7
          v8h h8 = *(const v8h*)(arow[mt] + kc * 32 + hi * 16 + 8);  // K = hi*16 + 8..15
#pragma unroll
          for (int i = 0; i < 8; i++) { av[i] = l8[i]; av[8 + i] = h8[i]; }
        } else {
#pragma unroll
          for (int i = 0; i < 16; i++) av[i] = (_Float16)0.f;
        }
        acc[mt] = __builtin_amdgcn_wmma_f32_16x16x32_f16(false, av, false, bv,
                                                         (short)0, acc[mt], false, false);
      }
    }
  }
  float bcol = bias ? bias[tileN * 16 + lo] : 0.f;
#pragma unroll
  for (int mt = 0; mt < MT; ++mt) {
#pragma unroll
    for (int r = 0; r < 8; ++r) {                  // D layout: VGPR r, row = r + hi*8, col = lo
      int row = (tileM0 + mt) * 16 + r + hi * 8;
      size_t o = (size_t)row * Cout + tileN * 16 + lo;
      float v = acc[mt][r] + bcol;
      if (resid) v += resid[o];
      if (relu) v = fmaxf(v, 0.f);
      out[o] = v;
    }
  }
}

// duration predictor head: conv k=1, 128 -> 1
__global__ void dp_out_k(const float* __restrict__ x, const float* __restrict__ w,
                         const float* __restrict__ bias, float* __restrict__ out, int rows) {
  int row = blockIdx.x, lane = threadIdx.x;
  float s = 0.f;
  for (int c = lane; c < CCH; c += 32) s += x[(size_t)row * CCH + c] * w[c];
  for (int off = 16; off; off >>= 1) s += __shfl_down(s, off, 32);
  if (lane == 0) out[row] = s + bias[0];
}

// masked inclusive cumsum of durations per batch item (P=256)
__global__ void cumsum_k(const int* __restrict__ dur, const int* __restrict__ plen,
                         int* __restrict__ cums) {
  __shared__ int s[256];
  int b = blockIdx.x, p = threadIdx.x;
  int d = (p < plen[b]) ? dur[b * 256 + p] : 0;
  s[p] = d; __syncthreads();
  for (int off = 1; off < 256; off <<= 1) {
    int v = 0;
    if (p >= off) v = s[p - off];
    __syncthreads();
    s[p] += v;
    __syncthreads();
  }
  cums[b * 256 + p] = s[p];
}

// expand_enc: searchsorted(right) gather + positional encoding + mask; writes two copies
__global__ void expand_k(const float* __restrict__ enc, const int* __restrict__ cums,
                         float* __restrict__ x0, float* __restrict__ x1) {
  const int P = 256, TOUT = 2048;
  int r = blockIdx.x;
  int b = r / TOUT, t = r % TOUT;
  const int* c = cums + b * P;
  int loI = 0, hiI = P;
  while (loI < hiI) { int mid = (loI + hiI) >> 1; if (c[mid] <= t) loI = mid + 1; else hiI = mid; }
  int idx = loI < P ? loI : P - 1;
  float mask = (t < c[P - 1]) ? 1.f : 0.f;
  int ch = threadIdx.x;
  float ang = (float)t * powf(10000.f, -(float)(ch & ~1) / (float)CCH);
  float pe = (ch & 1) ? cosf(ang) : sinf(ang);
  float v = (enc[((size_t)(b * P + idx)) * CCH + ch] + pe) * mask;
  size_t o = (size_t)r * CCH + ch;
  x0[o] = v; x1[o] = v;
}

// ---------------- orchestration ----------------

extern "C" void kernel_launch(void* const* d_in, const int* in_sizes, int n_in,
                              void* d_out, int out_size, void* d_ws, size_t ws_size,
                              hipStream_t stream) {
  (void)in_sizes; (void)n_in; (void)out_size; (void)ws_size;
  const int B = 16, P = 256, TOUT = 2048, OUTC = 80;
  const int encRows = B * P;       // 4096
  const int decRows = B * TOUT;    // 32768

  const int* phon = (const int*)d_in[0];
  const int* plen = (const int*)d_in[1];
  const int* dur  = (const int*)d_in[2];

  // --- parameter leaf walk: top-level in dict order, params jax-tree-flattened (sorted keys) ---
  struct CV { const float* b; const float* w; };
  struct BN { const float* be; const float* g; };
  int li = 3;
  auto takeBN = [&]() { BN p{ (const float*)d_in[li], (const float*)d_in[li + 1] }; li += 2; return p; };
  auto takeCV = [&]() { CV p{ (const float*)d_in[li], (const float*)d_in[li + 1] }; li += 2; return p; };

  BN dec_bn[17][2]; CV dec_cv[17][2];
  for (int i = 0; i < 17; i++) for (int j = 0; j < 2; j++) { dec_bn[i][j] = takeBN(); dec_cv[i][j] = takeCV(); }
  CV dec_post1 = takeCV();
  CV dec_post2 = takeCV();
  BN dp2_bn[2]; CV dp2_cv[2];
  for (int j = 0; j < 2; j++) { dp2_bn[j] = takeBN(); dp2_cv[j] = takeCV(); }
  BN dp_bn[3]; CV dp_cv[3];
  for (int i = 0; i < 3; i++) { dp_bn[i] = takeBN(); dp_cv[i] = takeCV(); }
  CV dp_head = takeCV();
  const float* emb = (const float*)d_in[li++];
  BN enc_bn[13][2]; CV enc_cv[13][2];
  for (int i = 0; i < 13; i++) for (int j = 0; j < 2; j++) { enc_bn[i][j] = takeBN(); enc_cv[i][j] = takeCV(); }
  CV enc_post1 = takeCV();
  CV enc_post2 = takeCV();
  BN enc_post2_bn = takeBN();
  CV enc_prenet = takeCV();

  // --- workspace layout ---
  char* ws = (char*)d_ws;
  float* X0 = (float*)ws;
  float* X  = X0 + (size_t)decRows * CCH;
  float* H  = X  + (size_t)decRows * CCH;
  _Float16* HF = (_Float16*)(H + (size_t)decRows * CCH);
  _Float16* WP = HF + (size_t)decRows * CCH;
  float* STATS = (float*)(WP + 4 * 128 * 128);
  int* CUMS = (int*)(STATS + 2 * CCH);

  float* E   = X0;                              // encoder prenet output (skip source)
  float* Xe  = X;                               // encoder working buffer
  float* He  = H;                               // encoder inter-conv temp
  float* ENC = H + (size_t)encRows * CCH;       // encoder output
  float* XDP = X0;                              // duration predictor working (reuses E)
  float* OUT = (float*)d_out;

  // --- launch helpers ---
  auto conv = [&](const _Float16* a, CV cv, const float* resid, float* out,
                  int Tloc, int Cout, int KW, int dil, int relu) {
    int pl = (dil * (KW - 1)) / 2;
    int total = KW * (CCH / 32) * (Cout / 16) * 512;
    pack_w_k<<<dim3((total + 255) / 256), dim3(256), 0, stream>>>(cv.w, WP, KW, CCH, Cout);
    int rows = B * Tloc;
    dim3 gy((Cout / 16 + 7) / 8);
    if (Tloc == TOUT) {
      dim3 grid(rows / (16 * 4), gy.x);
      conv_wmma_k<4, 11><<<grid, dim3(256), 0, stream>>>(a, WP, cv.b, resid, out,
                                                         Cout, KW, dil, pl, relu);
    } else {
      dim3 grid(rows / (16 * 2), gy.x);
      conv_wmma_k<2, 8><<<grid, dim3(256), 0, stream>>>(a, WP, cv.b, resid, out,
                                                        Cout, KW, dil, pl, relu);
    }
  };
  auto tof16 = [&](const float* xin, int rows) {
    bnrelu_k<<<dim3((rows * CCH + 255) / 256), dim3(256), 0, stream>>>(
        xin, STATS, (const float*)nullptr, (const float*)nullptr, HF, rows * CCH, 2);
  };
  auto bnconv = [&](const float* xin, BN bn, CV cv, const float* resid, float* out,
                    int Tloc, int KW, int dil) {
    int rows = B * Tloc;
    bn_stats_k<<<dim3(CCH), dim3(256), 0, stream>>>(xin, STATS, rows, 0);
    bnrelu_k<<<dim3((rows * CCH + 255) / 256), dim3(256), 0, stream>>>(
        xin, STATS, bn.g, bn.be, HF, rows * CCH, 0);
    conv(HF, cv, resid, out, Tloc, CCH, KW, dil, 0);
  };

  // ================= encoder =================
  gather_emb_k<<<dim3((encRows * CCH + 255) / 256), dim3(256), 0, stream>>>(phon, emb, HF, encRows);
  conv(HF, enc_prenet, nullptr, E, P, CCH, 1, 1, 1);               // relu(prenet)
  hipMemcpyAsync(Xe, E, (size_t)encRows * CCH * sizeof(float), hipMemcpyDeviceToDevice, stream);

  static const int ENC_DILS[13] = {1,2,4, 1,2,4, 1,2,4, 1,2,4, 1};
  for (int i = 0; i < 13; i++) {
    bnconv(Xe, enc_bn[i][0], enc_cv[i][0], nullptr, He, P, 4, ENC_DILS[i]);
    bnconv(He, enc_bn[i][1], enc_cv[i][1], Xe, Xe, P, 4, ENC_DILS[i]);
  }
  tof16(Xe, encRows);
  conv(HF, enc_post1, E, Xe, P, CCH, 1, 1, 0);                     // post1 + emb skip
  bn_stats_k<<<dim3(CCH), dim3(256), 0, stream>>>(Xe, STATS, encRows, 1);
  bnrelu_k<<<dim3((encRows * CCH + 255) / 256), dim3(256), 0, stream>>>(
      Xe, STATS, enc_post2_bn.g, enc_post2_bn.be, HF, encRows * CCH, 1);
  conv(HF, enc_post2, nullptr, ENC, P, CCH, 1, 1, 0);              // -> ENC

  // ================= duration predictor =================
  static const int DPK[3] = {4, 3, 1};
  bnconv(ENC, dp_bn[0], dp_cv[0], ENC, XDP, P, DPK[0], 1);
  bnconv(XDP, dp_bn[1], dp_cv[1], XDP, XDP, P, DPK[1], 1);
  bnconv(XDP, dp_bn[2], dp_cv[2], XDP, XDP, P, DPK[2], 1);
  dp_out_k<<<dim3(encRows), dim3(32), 0, stream>>>(XDP, dp_head.w, dp_head.b,
                                                   OUT + (size_t)decRows * OUTC, encRows);

  // ================= expand =================
  cumsum_k<<<dim3(B), dim3(P), 0, stream>>>(dur, plen, CUMS);
  expand_k<<<dim3(decRows), dim3(CCH), 0, stream>>>(ENC, CUMS, X0, X);

  // ================= decoder =================
  static const int DEC_DILS[17] = {1,2,4,8, 1,2,4,8, 1,2,4,8, 1,2,4,8, 1};
  for (int i = 0; i < 17; i++) {
    bnconv(X, dec_bn[i][0], dec_cv[i][0], nullptr, H, TOUT, 4, DEC_DILS[i]);
    bnconv(H, dec_bn[i][1], dec_cv[i][1], X, X, TOUT, 4, DEC_DILS[i]);
  }
  tof16(X, decRows);
  conv(HF, dec_post1, X0, X, TOUT, CCH, 1, 1, 0);                  // post1 + expanded skip
  bnconv(X, dp2_bn[0], dp2_cv[0], nullptr, H, TOUT, 4, 1);         // post2 res block
  bnconv(H, dp2_bn[1], dp2_cv[1], X, X, TOUT, 4, 1);
  tof16(X, decRows);
  conv(HF, dec_post2, nullptr, OUT, TOUT, OUTC, 1, 1, 0);          // final 128 -> 80 mel
}
